// roPEAttentionHead_74921409511570
// MI455X (gfx1250) — matrix operand order repaired and checked by reading, not compiled, not run on hardware
//
#include <hip/hip_runtime.h>
#include <hip/hip_bf16.h>

#define B_DIM 8
#define C_DIM 2048
#define D_DIM 1024
// ln(10000)/D for RoPE inv_freq = exp(-(2j)/D * ln(10000))
#define ROPE_LN_OVER_D (9.210340371976184f / 1024.0f)

typedef __attribute__((ext_vector_type(16))) __bf16 v16bf;
typedef __attribute__((ext_vector_type(8)))  float  v8f;

union FragBF { v16bf v; unsigned int u[8]; };

__device__ inline v8f zero8() {
  v8f z = {0.f, 0.f, 0.f, 0.f, 0.f, 0.f, 0.f, 0.f};
  return z;
}

// fp32 -> bf16 round-to-nearest-even (only in O(N^2) epilogues / convert pass)
__device__ inline unsigned short f2bf(float f) {
  union { float f; unsigned u; } c; c.f = f;
  unsigned u = c.u;
  u += 0x7FFFu + ((u >> 16) & 1u);
  return (unsigned short)(u >> 16);
}
__device__ inline unsigned packbf2(float lo, float hi) {
  return (unsigned)f2bf(lo) | ((unsigned)f2bf(hi) << 16);
}

// ---- Fragment loaders (CDNA5 16x16x32 bf16 WMMA layouts, ISA 7.12.2) ----
// A: 16x32 (MxK). lane(0-15)=row M; two contiguous 16B runs per lane -> b128 loads.
__device__ inline void load_a_bf16(FragBF& A, const unsigned short* __restrict__ src,
                                   long row0, int k0, int ld, int lane) {
  int half = lane >> 4, m = lane & 15;
  const unsigned short* p = src + (row0 + m) * (long)ld + k0 + half * 8;
#pragma unroll
  for (int i = 0; i < 4; ++i) A.u[i] = *(const unsigned int*)(p + 2 * i);
#pragma unroll
  for (int i = 0; i < 4; ++i) A.u[4 + i] = *(const unsigned int*)(p + 16 + 2 * i);
}
// B: 32x16 (KxN), B_wmma[k,n] = src[(nrow0+n)*ld + k0+k]  (K contiguous in memory)
//   Two contiguous 16B runs per lane (same k order as A) -> b128 loads.
__device__ inline void load_b_kt_bf16(FragBF& Bf, const unsigned short* __restrict__ src,
                                      long nrow0, int k0, int ld, int lane) {
  int half = lane >> 4, n = lane & 15;
  const unsigned short* p = src + (nrow0 + n) * (long)ld + k0 + half * 8;
#pragma unroll
  for (int j = 0; j < 4; ++j) Bf.u[j] = *(const unsigned int*)(p + 2 * j);
#pragma unroll
  for (int j = 0; j < 4; ++j) Bf.u[4 + j] = *(const unsigned int*)(p + 16 + 2 * j);
}

#define WMMA_BF16(ACC, A, Bf) \
  (ACC) = __builtin_amdgcn_wmma_f32_16x16x32_bf16(false, (A).v, false, (Bf).v, (short)0, (ACC), false, false)

// ============ Kernel 0: bulk fp32 -> bf16 conversion (O(N^2) pass) ============
__global__ __launch_bounds__(256) void cvt_bf16_kernel(const float* __restrict__ src,
                                                       unsigned short* __restrict__ dst,
                                                       long n) {
  long i = ((long)blockIdx.x * blockDim.x + threadIdx.x) * 8;
  if (i >= n) return;
#pragma unroll
  for (int j = 0; j < 8; j += 2) {
    float2 f = *(const float2*)(src + i + j);
    *(unsigned int*)(dst + i + j) = packbf2(f.x, f.y);
  }
}

// ============ Kernel 1: Q/K/V projection + bias + fused RoPE ============
// One wave: 16x64 output tile of q, k AND v; A fragment reused for 12 WMMAs / K-step.
// Bq/Bk/Bv get distinct registers so loads overlap preceding WMMAs (partial waits).
// V is stored TRANSPOSED [B, D, C] so the PV GEMM gets contiguous-K fragments.
__global__ __launch_bounds__(256) void qkv_rope_kernel(
    const unsigned short* __restrict__ xbf,
    const unsigned short* __restrict__ wqbf, const float* __restrict__ bq,
    const unsigned short* __restrict__ wkbf, const float* __restrict__ bk,
    const unsigned short* __restrict__ wvbf, const float* __restrict__ bv,
    unsigned short* __restrict__ qbf, unsigned short* __restrict__ kbf,
    unsigned short* __restrict__ vtbf) {
  int wave = (int)((blockIdx.x * blockDim.x + threadIdx.x) >> 5);
  int lane = threadIdx.x & 31;
  const int ngroups = D_DIM / 64;  // 16
  int mtile = wave / ngroups;      // 0..1023
  int ng = wave % ngroups;
  long m0 = (long)mtile * 16;
  int n0 = ng * 64;

  v8f accq[4], acck[4], accv[4];
#pragma unroll
  for (int t = 0; t < 4; ++t) { accq[t] = zero8(); acck[t] = zero8(); accv[t] = zero8(); }

  for (int k0 = 0; k0 < D_DIM; k0 += 32) {
    FragBF A;
    load_a_bf16(A, xbf, m0, k0, D_DIM, lane);
#pragma unroll
    for (int t = 0; t < 4; ++t) {
      FragBF Bq, Bk, Bv;  // distinct regs -> loads for later WMMAs overlap earlier WMMAs
      load_b_kt_bf16(Bq, wqbf, n0 + t * 16, k0, D_DIM, lane);
      load_b_kt_bf16(Bk, wkbf, n0 + t * 16, k0, D_DIM, lane);
      load_b_kt_bf16(Bv, wvbf, n0 + t * 16, k0, D_DIM, lane);
      WMMA_BF16(accq[t], A, Bq);
      WMMA_BF16(acck[t], A, Bk);
      WMMA_BF16(accv[t], A, Bv);
    }
  }

  // Epilogue: bias + RoPE (pair partner is in adjacent lane: lane parity == dim parity)
  int half = lane >> 4, nn = lane & 15;
#pragma unroll
  for (int t = 0; t < 4; ++t) {
#pragma unroll
    for (int r = 0; r < 8; ++r) {
      int m = r + 8 * half;
      long row = m0 + m;
      int b = (int)(row >> 11);            // row / C
      int pos = (int)(row & (C_DIM - 1));  // row % C
      int d = n0 + t * 16 + nn;
      float vq = accq[t][r] + bq[d];
      float vk = acck[t][r] + bk[d];
      float vv = accv[t][r] + bv[d];
      float freq = __expf(-(float)(d & ~1) * ROPE_LN_OVER_D);
      float ang = (float)pos * freq;
      float cs = __cosf(ang), sn = __sinf(ang);
      float pq = __shfl_xor(vq, 1, 32);
      float pk = __shfl_xor(vk, 1, 32);
      float oq = (d & 1) ? (pq * sn + vq * cs) : (vq * cs - pq * sn);
      float ok = (d & 1) ? (pk * sn + vk * cs) : (vk * cs - pk * sn);
      long idx = row * D_DIM + d;
      qbf[idx] = f2bf(oq);
      kbf[idx] = f2bf(ok);
      vtbf[((long)b * D_DIM + d) * C_DIM + pos] = f2bf(vv);  // transposed V
    }
  }
}

// ============ Kernel 2: S = scale * Qrot @ Krot^T, causal mask ============
// One wave: 16 queries x 64 keys (4 tiles) -> Q fragment amortized over 4 WMMAs.
// nact (live tiles vs causal diagonal) is wave-uniform: EXEC all-ones around WMMA.
__global__ __launch_bounds__(256) void scores_kernel(
    const unsigned short* __restrict__ qbf, const unsigned short* __restrict__ kbf,
    float* __restrict__ S) {
  int wave = (int)((blockIdx.x * blockDim.x + threadIdx.x) >> 5);
  int lane = threadIdx.x & 31;
  const int qtiles = C_DIM / 16;   // 128
  const int kgroups = C_DIM / 64;  // 32
  int kg = wave % kgroups;
  int qt = (wave / kgroups) % qtiles;
  int b = wave / (kgroups * qtiles);
  int half = lane >> 4, nn = lane & 15;
  float* Sb = S + (((long)b * C_DIM + qt * 16) * (long)C_DIM) + kg * 64;

  // tile t (keys kg*64+t*16 ..+15) is live iff kg*4 + t <= qt
  int nact = qt - kg * 4 + 1;
  if (nact > 4) nact = 4;

  v8f acc[4];
#pragma unroll
  for (int t = 0; t < 4; ++t) acc[t] = zero8();

  if (nact > 0) {
    long qrow0 = (long)b * C_DIM + qt * 16;
    long krow0 = (long)b * C_DIM + kg * 64;
    for (int k0 = 0; k0 < D_DIM; k0 += 32) {
      FragBF A;
      load_a_bf16(A, qbf, qrow0, k0, D_DIM, lane);
      FragBF Bf[4];
#pragma unroll
      for (int t = 0; t < 4; ++t)
        if (t < nact) load_b_kt_bf16(Bf[t], kbf, krow0 + t * 16, k0, D_DIM, lane);
#pragma unroll
      for (int t = 0; t < 4; ++t)
        if (t < nact) WMMA_BF16(acc[t], A, Bf[t]);
    }
  }

  const float scale = 0.03125f;  // 1/sqrt(1024)
#pragma unroll
  for (int t = 0; t < 4; ++t) {
#pragma unroll
    for (int r = 0; r < 8; ++r) {
      int m = r + 8 * half;
      int qi = qt * 16 + m;
      int kj = kg * 64 + t * 16 + nn;
      float s = -__builtin_inff();
      if (t < nact) {
        s = acc[t][r] * scale;
        if (kj > qi) s = -__builtin_inff();
      }
      Sb[(long)m * C_DIM + t * 16 + nn] = s;
    }
  }
}

// ============ Kernel 3: row softmax, P (bf16) = softmax(S) ============
__global__ __launch_bounds__(256) void softmax_kernel(const float* __restrict__ S,
                                                      unsigned short* __restrict__ P) {
  long row = blockIdx.x;  // 0 .. B*C-1
  const float* s = S + row * (long)C_DIM;
  unsigned short* p = P + row * (long)C_DIM;
  __shared__ float red[256];
  int tid = threadIdx.x;
  float mx = -__builtin_inff();
  for (int j = tid; j < C_DIM; j += 256) mx = fmaxf(mx, s[j]);
  red[tid] = mx;
  __syncthreads();
  for (int off = 128; off > 0; off >>= 1) {
    if (tid < off) red[tid] = fmaxf(red[tid], red[tid + off]);
    __syncthreads();
  }
  mx = red[0];
  __syncthreads();
  float sum = 0.f;
  for (int j = tid; j < C_DIM; j += 256) sum += __expf(s[j] - mx);
  red[tid] = sum;
  __syncthreads();
  for (int off = 128; off > 0; off >>= 1) {
    if (tid < off) red[tid] += red[tid + off];
    __syncthreads();
  }
  float inv = 1.f / red[0];
  for (int j = tid; j < C_DIM; j += 256) p[j] = f2bf(__expf(s[j] - mx) * inv);
}

// ============ Kernel 4: O = P @ V (per batch), causal-truncated K loop ============
// V is pre-transposed [B, D, C], so B fragments are contiguous-K dword loads.
__global__ __launch_bounds__(256) void pv_kernel(const unsigned short* __restrict__ P,
                                                 const unsigned short* __restrict__ vtbf,
                                                 float* __restrict__ out) {
  int wave = (int)((blockIdx.x * blockDim.x + threadIdx.x) >> 5);
  int lane = threadIdx.x & 31;
  const int ngroups = D_DIM / 64;  // 16
  const int qtiles = C_DIM / 16;   // 128
  int ng = wave % ngroups;
  int qt = (wave / ngroups) % qtiles;
  int b = wave / (ngroups * qtiles);
  int n0 = ng * 64;
  long prow0 = (long)b * C_DIM + qt * 16;
  long vtrow0 = (long)b * D_DIM;  // rows of V^T are output dims

  v8f acc[4];
#pragma unroll
  for (int t = 0; t < 4; ++t) acc[t] = zero8();

  int kmax = (qt + 1) * 16;  // P is exactly 0 beyond causal boundary
  for (int k0 = 0; k0 < kmax; k0 += 32) {
    FragBF A;
    load_a_bf16(A, P, prow0, k0, C_DIM, lane);
    FragBF Bf[4];
#pragma unroll
    for (int t = 0; t < 4; ++t)
      load_b_kt_bf16(Bf[t], vtbf, vtrow0 + n0 + t * 16, k0, C_DIM, lane);
#pragma unroll
    for (int t = 0; t < 4; ++t) WMMA_BF16(acc[t], A, Bf[t]);
  }

  int half = lane >> 4, nn = lane & 15;
#pragma unroll
  for (int t = 0; t < 4; ++t) {
#pragma unroll
    for (int r = 0; r < 8; ++r) {
      int m = r + 8 * half;
      long row = (long)b * C_DIM + qt * 16 + m;
      out[row * D_DIM + n0 + t * 16 + nn] = acc[t][r];
    }
  }
}

extern "C" void kernel_launch(void* const* d_in, const int* in_sizes, int n_in,
                              void* d_out, int out_size, void* d_ws, size_t ws_size,
                              hipStream_t stream) {
  const float* x = (const float*)d_in[0];
  const float* Wq = (const float*)d_in[1];
  const float* bq = (const float*)d_in[2];
  const float* Wk = (const float*)d_in[3];
  const float* bk = (const float*)d_in[4];
  const float* Wv = (const float*)d_in[5];
  const float* bv = (const float*)d_in[6];
  float* out = (float*)d_out;

  // Workspace layout
  char* ws = (char*)d_ws;
  const size_t nBD = (size_t)B_DIM * C_DIM * D_DIM;   // 16M elems
  const size_t nBCC = (size_t)B_DIM * C_DIM * C_DIM;  // 33.5M elems
  const size_t nDD = (size_t)D_DIM * D_DIM;           // 1M elems
  unsigned short* qbf  = (unsigned short*)ws; ws += nBD * 2;
  unsigned short* kbf  = (unsigned short*)ws; ws += nBD * 2;
  unsigned short* vtbf = (unsigned short*)ws; ws += nBD * 2;
  float* S = (float*)ws;                      ws += nBCC * 4;
  unsigned short* P = (unsigned short*)ws;    ws += nBCC * 2;
  unsigned short* xbf  = (unsigned short*)ws; ws += nBD * 2;
  unsigned short* wqbf = (unsigned short*)ws; ws += nDD * 2;
  unsigned short* wkbf = (unsigned short*)ws; ws += nDD * 2;
  unsigned short* wvbf = (unsigned short*)ws; ws += nDD * 2;

  // k0: bulk bf16 conversion (8 elems/thread, 2048 elems/block)
  cvt_bf16_kernel<<<(int)(nBD / 2048), 256, 0, stream>>>(x, xbf, (long)nBD);
  cvt_bf16_kernel<<<(int)(nDD / 2048), 256, 0, stream>>>(Wq, wqbf, (long)nDD);
  cvt_bf16_kernel<<<(int)(nDD / 2048), 256, 0, stream>>>(Wk, wkbf, (long)nDD);
  cvt_bf16_kernel<<<(int)(nDD / 2048), 256, 0, stream>>>(Wv, wvbf, (long)nDD);

  // k1: (16384/16 m-tiles) x (1024/64 n-groups) waves = 16384 waves
  qkv_rope_kernel<<<2048, 256, 0, stream>>>(xbf, wqbf, bq, wkbf, bk, wvbf, bv,
                                            qbf, kbf, vtbf);
  // k2: 8 * 128 qtiles * 32 kgroups waves = 32768 waves
  scores_kernel<<<4096, 256, 0, stream>>>(qbf, kbf, S);
  // k3: one block per score row
  softmax_kernel<<<16384, 256, 0, stream>>>(S, P);
  // k4: 8 * 128 * 16 waves
  pv_kernel<<<2048, 256, 0, stream>>>(P, vtbf, out);
}
